// MAPLoss_85461259255891
// MI455X (gfx1250) — compile-verified, matrix operand order checked by reading, not run on hardware
//
#include <hip/hip_runtime.h>
#include <hip/hip_bf16.h>

typedef __attribute__((ext_vector_type(16))) _Float16 v16h;
typedef __attribute__((ext_vector_type(8)))  _Float16 v8h;
typedef __attribute__((ext_vector_type(8)))  float    v8f;

#define NROW 512
#define DDIM 128
#define EPSV 1e-8f
// 1/K * log2(e) : sigmoid((a-b)/K) = 1 / (1 + 2^((b-a)*KLOG2E))
#define KLOG2E 1442.6950408889634f

// ---------------------------------------------------------------------------
// 1) Row L2-normalize, emit f16 normalized embeddings (WMMA operand precision)
//    One wave (32 lanes) per row; 4 elems/lane; shfl_xor wave reduction.
// ---------------------------------------------------------------------------
__global__ void norm_kernel(const float* __restrict__ x, _Float16* __restrict__ xn) {
    int row  = blockIdx.x;
    int lane = threadIdx.x;          // 0..31 (wave32)
    float v[4];
    float ss = 0.0f;
#pragma unroll
    for (int k = 0; k < 4; ++k) {
        v[k] = x[row * DDIM + lane + 32 * k];
        ss = fmaf(v[k], v[k], ss);
    }
#pragma unroll
    for (int m = 16; m >= 1; m >>= 1) ss += __shfl_xor(ss, m, 32);
    float inv = 1.0f / fmaxf(sqrtf(ss), EPSV);
#pragma unroll
    for (int k = 0; k < 4; ++k)
        xn[row * DDIM + lane + 32 * k] = (_Float16)(v[k] * inv);
}

// ---------------------------------------------------------------------------
// 2) sm = xn @ xn^T via v_wmma_f32_16x16x32_f16.
//    One wave per 16x16 tile; 32x32 tiles; 4 waves/block -> 256 blocks.
//    Because B = xn^T, the A and B per-lane load patterns are identical on
//    row-major xn:
//      lane L: g = L>>4, m = L&15
//      A half a[2v],a[2v+1]: K = (v<4 ? 2v : 16+2(v-4)) + 8g  (+kk)
//    -> two contiguous v8h (16B) loads per operand per K-step.
//    C/D f32 layout: lane L, vgpr v -> (M = v + 8g, N = m).
// ---------------------------------------------------------------------------
__global__ void gemm_wmma_kernel(const _Float16* __restrict__ xn,
                                 float* __restrict__ sm) {
    int wave = threadIdx.x >> 5;
    int lane = threadIdx.x & 31;
    int tile = blockIdx.x * 4 + wave;      // 0..1023
    int ti = tile >> 5;                    // row tile
    int tj = tile & 31;                    // col tile
    int g = lane >> 4;
    int m = lane & 15;

    const _Float16* arow = xn + (ti * 16 + m) * DDIM;
    const _Float16* brow = xn + (tj * 16 + m) * DDIM;

    v8f c = {};
#pragma unroll
    for (int kk = 0; kk < DDIM; kk += 32) {
        v8h alo = *(const v8h*)(arow + kk + 8 * g);
        v8h ahi = *(const v8h*)(arow + kk + 16 + 8 * g);
        v8h blo = *(const v8h*)(brow + kk + 8 * g);
        v8h bhi = *(const v8h*)(brow + kk + 16 + 8 * g);
        v16h a, b;
#pragma unroll
        for (int t = 0; t < 8; ++t) {
            a[t] = alo[t]; a[8 + t] = ahi[t];
            b[t] = blo[t]; b[8 + t] = bhi[t];
        }
        // (neg_a, A, neg_b, B, c_mod, C, reuse_a, reuse_b)
        c = __builtin_amdgcn_wmma_f32_16x16x32_f16(
                false, a, false, b, (short)0, c, false, false);
    }
#pragma unroll
    for (int v = 0; v < 8; ++v) {
        int rr = ti * 16 + v + 8 * g;
        int cc = tj * 16 + m;
        sm[rr * NROW + cc] = c[v];
    }
}

// ---------------------------------------------------------------------------
// 3) Per-row MAP cube: one 256-thread block per row i.
//    LDS holds interleaved (r, gt) pairs -> one ds_load_b64 per sigmoid.
//    Diagonal handled branchlessly: r[i] = -1e30 makes its sigmoid term
//    exactly 0 (2^(+huge) -> inf -> rcp -> 0) and gt[i] = 0.
//    Sigmoid fused into: x = fma(-KLOG2E, r_l, rjC); s = rcp(1 + exp2(x))
//    -> per iteration: ds_b64 + fma + v_exp_f32 + add + v_rcp_f32 + add + fma.
// ---------------------------------------------------------------------------
__global__ void maprow_kernel(const float* __restrict__ sm,
                              const int* __restrict__ tgt,
                              float* __restrict__ ap,
                              float* __restrict__ npos) {
    __shared__ float2 rgS[NROW];     // .x = similarity, .y = gt flag
    __shared__ float redA[256];
    __shared__ float redN[256];

    int i   = blockIdx.x;
    int tid = threadIdx.x;
    int ci  = tgt[i];

    for (int c = tid; c < NROW; c += 256) {
        float rv = (c == i) ? -1.0e30f : sm[i * NROW + c];
        float gv = (c != i && tgt[c] == ci) ? 1.0f : 0.0f;
        rgS[c] = make_float2(rv, gv);
    }
    __syncthreads();

    float apLoc = 0.0f, npLoc = 0.0f;
    for (int j = tid; j < NROW; j += 256) {
        if (j == i) continue;
        float rjC = rgS[j].x * KLOG2E;
        float s1 = 0.0f, sg = 0.0f;
#pragma unroll 4
        for (int l = 0; l < NROW; ++l) {
            float2 rg = rgS[l];
            float xx = fmaf(-KLOG2E, rg.x, rjC);            // (r_j - r_l)/K * log2e
            float e  = __builtin_amdgcn_exp2f(xx);          // v_exp_f32 (base-2)
            float s  = __builtin_amdgcn_rcpf(1.0f + e);     // v_rcp_f32
            s1 += s;
            sg  = fmaf(rg.y, s, sg);
        }
        float prec = sg / (s1 + 0.5f);
        float gj = rgS[j].y;
        apLoc = fmaf(gj, prec, apLoc);
        npLoc += gj;
    }

    redA[tid] = apLoc;
    redN[tid] = npLoc;
    __syncthreads();
    for (int s = 128; s > 0; s >>= 1) {
        if (tid < s) { redA[tid] += redA[tid + s]; redN[tid] += redN[tid + s]; }
        __syncthreads();
    }
    if (tid == 0) { ap[i] = redA[0]; npos[i] = redN[0]; }
}

// ---------------------------------------------------------------------------
// 4) loss = 1 - sum(ap/np over valid rows) / n_valid
// ---------------------------------------------------------------------------
__global__ void finalize_kernel(const float* __restrict__ ap,
                                const float* __restrict__ npos,
                                float* __restrict__ out) {
    __shared__ float sA[NROW];
    __shared__ float sC[NROW];
    int t = threadIdx.x;
    float np = npos[t];
    bool valid = np > 0.0f;
    sA[t] = valid ? ap[t] / np : 0.0f;
    sC[t] = valid ? 1.0f : 0.0f;
    __syncthreads();
    for (int s = 256; s > 0; s >>= 1) {
        if (t < s) { sA[t] += sA[t + s]; sC[t] += sC[t + s]; }
        __syncthreads();
    }
    if (t == 0) out[0] = 1.0f - sA[0] / sC[0];
}

// ---------------------------------------------------------------------------
extern "C" void kernel_launch(void* const* d_in, const int* in_sizes, int n_in,
                              void* d_out, int out_size, void* d_ws, size_t ws_size,
                              hipStream_t stream) {
    const float* x   = (const float*)d_in[0];   // [512,128] f32
    const int*   tgt = (const int*)d_in[1];     // [512] int

    char* ws = (char*)d_ws;
    _Float16* xn = (_Float16*)ws;                               // 131072 B
    float*    sm = (float*)(ws + 131072);                       // 1 MB
    float*    ap = (float*)(ws + 131072 + 1048576);             // 2 KB
    float*    np = (float*)(ws + 131072 + 1048576 + 2048);      // 2 KB

    norm_kernel<<<NROW, 32, 0, stream>>>(x, xn);
    gemm_wmma_kernel<<<256, 128, 0, stream>>>(xn, sm);
    maprow_kernel<<<NROW, 256, 0, stream>>>(sm, tgt, ap, np);
    finalize_kernel<<<1, NROW, 0, stream>>>(ap, np, (float*)d_out);
}
